// InteractionHead_29703993819506
// MI455X (gfx1250) — compile-verified, matrix operand order checked by reading, not run on hardware
//
#include <hip/hip_runtime.h>
#include <hip/hip_bf16.h>
#include <stdint.h>

// CDNA5 (gfx1250) wave32 WMMA fragment types
typedef __attribute__((ext_vector_type(16))) _Float16 v16h;
typedef __attribute__((ext_vector_type(8)))  float    v8f;

#define NTHREADS 256
#define N_DET    1024
#define NS_      512
#define SCORE_T  0.2f
// IoU > 0.5  <=>  inter/(sum-inter) > 0.5  <=>  3*inter > sum

__global__ __launch_bounds__(NTHREADS)
void ih_nms_kernel(const float* __restrict__ subj_boxes,
                   const float* __restrict__ subj_scores,
                   const int*   __restrict__ subj_labels,
                   const float* __restrict__ obj_boxes,
                   const float* __restrict__ obj_scores,
                   const int*   __restrict__ obj_labels,
                   float* __restrict__ out, int B)
{
    __shared__ uint64_t s_key[N_DET];       // (score_bits<<32)|(~idx) sort keys
    __shared__ float4   s_box[N_DET];       // sorted boxes (xyxy)
    __shared__ float    s_score[N_DET];     // sorted scores
    __shared__ float    s_area[N_DET];      // sorted areas
    __shared__ int      s_lab[N_DET];       // low16 = label, bit16 = is_subject
    __shared__ uint32_t s_cand[16][64];     // per chunk: 16 rows x 64 j-groups (16 bits used/word)
    __shared__ uint32_t s_sup[64];          // global suppressed bitmask, 16 bits per word
    __shared__ uint32_t s_keptbits;         // in-chunk kept rows (broadcast)

    const int b    = blockIdx.x;
    const int tid  = threadIdx.x;
    const int lane = tid & 31;              // wave32
    const int wave = tid >> 5;              // 8 waves / block

    // Pull this batch's slices toward cache (global_prefetch_b8)
    __builtin_prefetch(subj_boxes + (size_t)b * NS_ * 4, 0, 1);
    __builtin_prefetch(obj_boxes  + (size_t)b * NS_ * 4, 0, 1);
    __builtin_prefetch(subj_scores + (size_t)b * NS_, 0, 1);
    __builtin_prefetch(obj_scores  + (size_t)b * NS_, 0, 1);

    // ---- 1) build sort keys: score descending, index ascending on ties ----
    for (int r = tid; r < N_DET; r += NTHREADS) {
        float sc = (r < NS_) ? subj_scores[b * NS_ + r]
                             : obj_scores[b * NS_ + (r - NS_)];
        uint32_t sbits = __float_as_uint(sc);   // scores in [0,1): bit-monotonic
        s_key[r] = ((uint64_t)sbits << 32) | (uint32_t)(0xFFFFFFFFu - (uint32_t)r);
    }

    // ---- 2) bitonic sort, descending, in LDS ----
    for (unsigned k = 2; k <= N_DET; k <<= 1) {
        for (unsigned j = k >> 1; j > 0; j >>= 1) {
            __syncthreads();
            for (unsigned i = tid; i < N_DET; i += NTHREADS) {
                unsigned ixj = i ^ j;
                if (ixj > i) {
                    uint64_t a = s_key[i], c2 = s_key[ixj];
                    bool desc = ((i & k) == 0);
                    if ((a < c2) == desc) { s_key[i] = c2; s_key[ixj] = a; }
                }
            }
        }
    }
    __syncthreads();

    // ---- 3) gather boxes/labels into sorted order (b128 loads) ----
    for (int r = tid; r < N_DET; r += NTHREADS) {
        uint64_t kk = s_key[r];
        uint32_t jo = 0xFFFFFFFFu - (uint32_t)kk;
        float    sc = __uint_as_float((uint32_t)(kk >> 32));
        const int    isub = (jo < NS_) ? 1 : 0;
        const uint32_t j2 = isub ? jo : (jo - NS_);
        const float4* bp4 = isub ? ((const float4*)subj_boxes + (size_t)b * NS_ + j2)
                                 : ((const float4*)obj_boxes  + (size_t)b * NS_ + j2);
        const int*    lp  = isub ? (subj_labels + b * NS_ + j2)
                                 : (obj_labels  + b * NS_ + j2);
        float4 bx = *bp4;
        int    lab = *lp;
        s_box[r]   = bx;
        s_score[r] = sc;
        s_lab[r]   = (lab & 0xFFFF) | (isub << 16);
        s_area[r]  = (bx.z - bx.x) * (bx.w - bx.y);  // class offsets cancel out
    }
    for (int w = tid; w < 64; w += NTHREADS) s_sup[w] = 0;
    __syncthreads();

    // ---- 4) blocked greedy NMS: 64 chunks of 16 rows ----
    for (int c = 0; c < 64; ++c) {
        const int  cb     = c << 4;
        const int  rowoff = (lane & 16) ? 8 : 0;  // C/D: VGPR r -> M=r (lanes 0-15), M=r+8 (16-31)
        const bool lo     = (lane < 16);

        // A fragment once per chunk: row M=lane holds [area_i, 1, 0...] along K
        // (16-bit A layout: lanes 0-15 carry K0..7,K16..23 -> elem0=K0, elem1=K1)
        const float ai = s_area[cb + (lane & 15)];
        v16h afrag = {};
        afrag[0] = lo ? (_Float16)ai   : (_Float16)0.0f;
        afrag[1] = lo ? (_Float16)1.0f : (_Float16)0.0f;

        // My 8 i-rows, loaded once per chunk (hoisted out of the tile loop)
        float4 bi[8]; int li[8];
#pragma unroll
        for (int r = 0; r < 8; ++r) {
            const int ig = cb + rowoff + r;
            bi[r] = s_box[ig];
            li[r] = s_lab[ig] & 0xFFFF;
        }

        // phase 1: candidate masks; lower-triangle tiles (t<c) are identically
        // zero via the j>i gate, so skip them entirely.
        for (int t = wave; t < 64; t += 8) {
            if (t < c) continue;
            const int jb = t << 4;
            const int jg = jb + (lane & 15);

            // B fragment: col N=lane holds [1, area_j, 0...] along K (lanes 0-15: elem0=K0, elem1=K1)
            const float aj = s_area[jg];
            v16h bfrag = {};
            bfrag[0] = lo ? (_Float16)1.0f : (_Float16)0.0f;
            bfrag[1] = lo ? (_Float16)aj   : (_Float16)0.0f;
            v8f acc = {};
            // D[i][j] = area_i + area_j on the matrix pipe; 16-bit WMMA is
            // TRANS-class and co-executes with the VALU min/max below.
            acc = __builtin_amdgcn_wmma_f32_16x16x32_f16(
                false, afrag, false, bfrag, (short)0, acc, false, false);

            const float4 bj = s_box[jg];
            const int    lj = s_lab[jg] & 0xFFFF;

            uint32_t m[8];
#pragma unroll
            for (int r = 0; r < 8; ++r) {
                const int ig = cb + rowoff + r;
                float lx = fmaxf(bi[r].x, bj.x), ly = fmaxf(bi[r].y, bj.y);
                float rx = fminf(bi[r].z, bj.z), ry = fminf(bi[r].w, bj.w);
                float iw = fmaxf(rx - lx, 0.0f), ih = fmaxf(ry - ly, 0.0f);
                float inter = iw * ih;
                // bitwise & (no short-circuit) -> predicated straightline code
                int cnd = (int)(jg > ig) & (int)(lj == li[r]) &
                          (int)(3.0f * inter > acc[r]);
                m[r] = __builtin_amdgcn_ballot_w32(cnd != 0);
            }
            if (lane == 0) {
#pragma unroll
                for (int r = 0; r < 8; ++r) {
                    s_cand[r][t]     = m[r] & 0xFFFFu;   // rows cb+r,   j in [jb, jb+16)
                    s_cand[r + 8][t] = m[r] >> 16;       // rows cb+r+8
                }
            }
        }
        __syncthreads();

        // phase 2: resolve sequential dependency inside the chunk (16 steps)
        if (tid == 0) {
            uint32_t supw = s_sup[c];   // word c holds in-chunk suppression bits
            uint32_t kb = 0;
            for (int ci = 0; ci < 16; ++ci) {
                if ((s_score[cb + ci] >= SCORE_T) && !((supw >> ci) & 1)) {
                    kb |= (1u << ci);
                    supw |= s_cand[ci][c];   // in-chunk suppression from this kept row
                }
            }
            s_sup[c] = supw;
            s_keptbits = kb;
        }
        __syncthreads();

        // phase 3: apply kept rows' masks across j-groups >= c in parallel
        if (tid >= c && tid < 64) {
            uint32_t a2 = s_sup[tid];
            uint32_t k2 = s_keptbits;
            while (k2) {
                int ci = __builtin_ctz(k2);
                k2 &= (k2 - 1);
                a2 |= s_cand[ci][tid];
            }
            s_sup[tid] = a2;
        }
        __syncthreads();
    }

    // ---- 5) select top-15 subjects / objects (rows already score-sorted) ----
    if (tid == 0) {
        const size_t offS = (size_t)B * 120;
        const size_t offL = offS + (size_t)B * 30;
        const size_t offN = offL + (size_t)B * 30;
        const size_t offV = offN + (size_t)B;
        float* outB = out + (size_t)b * 120;
        float* outS = out + offS + (size_t)b * 30;
        float* outL = out + offL + (size_t)b * 30;
        float* outN = out + offN + b;
        float* outV = out + offV + (size_t)b * 30;

        for (int s = 0; s < 30; ++s) {      // invalid slots: boxes*0, score*0, label -1, valid 0
            outB[s * 4 + 0] = 0.0f; outB[s * 4 + 1] = 0.0f;
            outB[s * 4 + 2] = 0.0f; outB[s * 4 + 3] = 0.0f;
            outS[s] = 0.0f; outL[s] = -1.0f; outV[s] = 0.0f;
        }
        int ns = 0, no_ = 0;
        for (int r = 0; r < N_DET; ++r) {
            if (ns >= 15 && no_ >= 15) break;
            bool kept = (s_score[r] >= SCORE_T) &&
                        !((s_sup[r >> 4] >> (r & 15)) & 1);
            if (!kept) continue;
            int slot;
            if (s_lab[r] & (1 << 16)) { if (ns  >= 15) continue; slot = ns++; }
            else                      { if (no_ >= 15) continue; slot = 15 + no_++; }
            float4 bx = s_box[r];
            outB[slot * 4 + 0] = bx.x; outB[slot * 4 + 1] = bx.y;
            outB[slot * 4 + 2] = bx.z; outB[slot * 4 + 3] = bx.w;
            outS[slot] = s_score[r];
            outL[slot] = (float)(s_lab[r] & 0xFFFF);
            outV[slot] = 1.0f;
        }
        outN[0] = (float)ns;
    }
}

extern "C" void kernel_launch(void* const* d_in, const int* in_sizes, int n_in,
                              void* d_out, int out_size, void* d_ws, size_t ws_size,
                              hipStream_t stream) {
    (void)n_in; (void)out_size; (void)d_ws; (void)ws_size;
    const float* sb = (const float*)d_in[0];
    const float* ss = (const float*)d_in[1];
    const int*   sl = (const int*)d_in[2];
    const float* ob = (const float*)d_in[3];
    const float* os = (const float*)d_in[4];
    const int*   ol = (const int*)d_in[5];
    float* out = (float*)d_out;
    int B = in_sizes[1] / NS_;   // subj_scores is [B,512]
    ih_nms_kernel<<<B, NTHREADS, 0, stream>>>(sb, ss, sl, ob, os, ol, out, B);
}
